// ContextualCritic_34720515621310
// MI455X (gfx1250) — matrix operand order repaired and checked by reading, compiled, not run on hardware
//
#include <hip/hip_runtime.h>
#include <hip/hip_bf16.h>

// ---------------------------------------------------------------------------
// CDNA5 (gfx1250) implicit-GEMM conv stack with v_wmma_f32_16x16x32_bf16 and
// async global->LDS weight staging (ASYNCcnt path). Memory-bound workload:
// wide-N tiling (NSUB sub-tiles per wave) keeps activation traffic at ~1x per
// layer (~2.3GB total vs 23.3TB/s). bf16 operands, f32 accumulate, fused
// bias+LeakyReLU epilogue.
// ---------------------------------------------------------------------------

typedef __attribute__((ext_vector_type(16))) __bf16 v16bf;
typedef __attribute__((ext_vector_type(8)))  float  v8f;

union FragBF {
    v16bf v;
    unsigned int u[8];
};

typedef __attribute__((address_space(3))) unsigned char lds_u8;

__device__ __forceinline__ unsigned lds_off_of(void* p) {
    return (unsigned)(unsigned long long)(lds_u8*)p;
}

__device__ __forceinline__ unsigned short f32_to_bf16(float f) {
    unsigned int u = __float_as_uint(f);
    unsigned int r = u + 0x7FFFu + ((u >> 16) & 1u);   // round-to-nearest-even
    return (unsigned short)(r >> 16);
}

__device__ __forceinline__ unsigned short load_bf16(const float* p, long long i) {
    return f32_to_bf16(p[i]);
}
__device__ __forceinline__ unsigned short load_bf16(const unsigned short* p, long long i) {
    return p[i];
}
__device__ __forceinline__ void store_act(unsigned short* p, long long i, float v) {
    p[i] = f32_to_bf16(v);
}
__device__ __forceinline__ void store_act(float* p, long long i, float v) {
    p[i] = v;
}

// ---------------------------------------------------------------------------
// One-shot weight conversion: OIHW f32 -> bf16 with K padded to mult. of 32
// (zero tail) so conv B-tiles are unconditional bulk async copies.
// ---------------------------------------------------------------------------
__global__ __launch_bounds__(256)
void convert_weights_kernel(const float* __restrict__ src, unsigned short* __restrict__ dst,
                            int Ktot, int Kpad, int total)
{
    int i = blockIdx.x * 256 + threadIdx.x;
    if (i >= total) return;
    int co = i / Kpad;
    int kk = i - co * Kpad;
    dst[i] = (kk < Ktot) ? f32_to_bf16(src[(long long)co * Ktot + kk]) : (unsigned short)0;
}

// ---------------------------------------------------------------------------
// conv 5x5 stride2 pad2 + bias + LeakyReLU(0.2), implicit GEMM:
//   M = Nimg*Hout*Wout, N = Cout, K = Cin*25 (padded to Kpad, KCAP = max)
// Block: 128 threads (4 waves). Tile: 64(M) x 64*NSUB(N), K chunk = 32.
// Wave w owns N sub-tiles {w, w+4, ...} (NSUB of them) x 4 M sub-tiles.
// B tile staged via global_load_async_to_lds_b128 (ASYNCcnt); K descriptors
// precomputed packed in LDS -> 2 barriers per K chunk.
// ---------------------------------------------------------------------------
template <typename Tin, typename Tout, int NSUB, int KCAP>
__global__ __launch_bounds__(128)
void conv5x5s2_lrelu_wmma(const Tin* __restrict__ X, const unsigned short* __restrict__ Wbf,
                          const float* __restrict__ Bias, Tout* __restrict__ Y,
                          int Cin, int Hin, int Win,
                          int Cout, int Hout, int Wout, int HWout,
                          int Ktot, int Kpad)
{
    const int tid  = threadIdx.x;
    const int lane = tid & 31;
    const int wv   = tid >> 5;        // wave 0..3
    const int hi   = (lane >> 4);     // half-wave select
    const int l16  = lane & 15;

    __shared__ unsigned short Abuf[64][34];          // [m][k], 17-dword rows
    __shared__ unsigned short Bbuf[64 * NSUB][40];   // [n][k], 80B rows (16B-aligned)
    __shared__ int rImg[64], rOy[64], rOx[64];
    __shared__ int kdesc[KCAP];                      // (ci<<6)|(ky<<3)|kx

    const int mBase = blockIdx.x * 64;
    const int nBase = blockIdx.y * (64 * NSUB);

    if (tid < 64) {
        int m   = mBase + tid;
        int img = m / HWout;
        int pix = m - img * HWout;
        int oy  = pix / Wout;
        rImg[tid] = img;
        rOy[tid]  = oy;
        rOx[tid]  = pix - oy * Wout;
    }
    for (int k = tid; k < Kpad; k += 128) {
        int ci = k / 25;
        int rr = k - ci * 25;
        int ky = rr / 5;
        int kx = rr - ky * 5;
        kdesc[k] = (ci << 6) | (ky << 3) | kx;
    }

    // per-thread constant pieces of the async B copy (2*NSUB x b128 per chunk)
    unsigned ldsB[2 * NSUB];
    const unsigned short* wrow[2 * NSUB];
#pragma unroll
    for (int i = 0; i < 2 * NSUB; ++i) {
        int id = tid + i * 128;
        int cl = id >> 2, in = id & 3;
        ldsB[i] = lds_off_of(&Bbuf[cl][in * 8]);
        wrow[i] = Wbf + (long long)(nBase + cl) * Kpad + in * 8;
    }

    v8f zero = {};
    v8f acc[4][NSUB];
#pragma unroll
    for (int s = 0; s < 4; ++s)
#pragma unroll
        for (int j = 0; j < NSUB; ++j) acc[s][j] = zero;

    for (int k0 = 0; k0 < Kpad; k0 += 32) {
        __syncthreads();                      // previous compute done; LDS reusable

        // ---- B tile: async bulk copy 64*NSUB(co) x 32(k) bf16 -> LDS ----
#pragma unroll
        for (int i = 0; i < 2 * NSUB; ++i) {
            unsigned long long ga = (unsigned long long)(wrow[i] + k0);
            asm volatile("global_load_async_to_lds_b128 %0, %1, off"
                         :: "v"(ldsB[i]), "v"(ga) : "memory");
        }

        // ---- A tile: 64(m) x 32(k) gathered im2col, bf16 into LDS ----
#pragma unroll
        for (int i = 0; i < 16; ++i) {
            int idx = tid + i * 128;          // 0..2047
            int ml  = idx >> 5;
            int kk  = idx & 31;
            unsigned short v = 0;
            if (k0 + kk < Ktot) {
                int kd = kdesc[k0 + kk];
                int iy = rOy[ml] * 2 + ((kd >> 3) & 7) - 2;
                int ix = rOx[ml] * 2 + (kd & 7) - 2;
                if ((unsigned)iy < (unsigned)Hin && (unsigned)ix < (unsigned)Win) {
                    long long off = ((long long)(rImg[ml] * Cin + (kd >> 6)) * Hin + iy) * Win + ix;
                    v = load_bf16(X, off);
                }
            }
            Abuf[ml][kk] = v;
        }

        asm volatile("s_wait_asynccnt 0x0" ::: "memory");   // B tile landed in LDS
        __syncthreads();

        // ---- fragments, contiguous k = 16*half + s in both A and B, then WMMA ----
        FragBF afr[4];
#pragma unroll
        for (int s = 0; s < 4; ++s) {
            const unsigned int* ap = (const unsigned int*)&Abuf[s * 16 + l16][0];
            int o0 = hi ? 8 : 0;  // lane<16: K 0..15; lane>=16: K 16..31
#pragma unroll
            for (int j = 0; j < 4; ++j) afr[s].u[j]     = ap[o0 + j];
#pragma unroll
            for (int j = 0; j < 4; ++j) afr[s].u[4 + j] = ap[o0 + 4 + j];
        }
#pragma unroll
        for (int j = 0; j < NSUB; ++j) {
            FragBF bfr;
            {
                // lane<16: col n=l16, K 0..15 ; lane>=16: col n=l16, K 16..31
                const unsigned int* bp =
                    (const unsigned int*)&Bbuf[(wv + 4 * j) * 16 + l16][hi ? 16 : 0];
#pragma unroll
                for (int q = 0; q < 8; ++q) bfr.u[q] = bp[q];
            }
#pragma unroll
            for (int s = 0; s < 4; ++s) {
                acc[s][j] = __builtin_amdgcn_wmma_f32_16x16x32_bf16(
                    false, afr[s].v, false, bfr.v, (short)0, acc[s][j], false, false);
            }
        }
    }

    // ---- epilogue: bias + LeakyReLU, scatter to NCHW output ----
#pragma unroll
    for (int j = 0; j < NSUB; ++j) {
        const int co = nBase + (wv + 4 * j) * 16 + l16;
        const float bv = Bias[co];
#pragma unroll
        for (int s = 0; s < 4; ++s) {
#pragma unroll
            for (int r = 0; r < 8; ++r) {
                int ml = s * 16 + r + (hi ? 8 : 0);
                float v = acc[s][j][r] + bv;
                v = (v >= 0.0f) ? v : 0.2f * v;
                long long off = ((long long)(rImg[ml] * Cout + co) * Hout + rOy[ml]) * Wout + rOx[ml];
                store_act(Y, off, v);
            }
        }
    }
}

// ---------------------------------------------------------------------------
// Segment counts + start offsets (seg is sorted, tiny problem: 2048 objects)
// ---------------------------------------------------------------------------
__global__ void seg_count_kernel(const int* __restrict__ seg, int nObj, int B,
                                 int* __restrict__ counts, int* __restrict__ starts)
{
    for (int b = threadIdx.x; b < B; b += blockDim.x) {
        int c = 0;
        for (int i = 0; i < nObj; ++i) c += (seg[i] == b) ? 1 : 0;
        counts[b] = c;
    }
    __syncthreads();
    for (int b = threadIdx.x; b < B; b += blockDim.x) {
        int s = 0;
        for (int j = 0; j < b; ++j) s += counts[j];
        starts[b] = s;
    }
}

// ---------------------------------------------------------------------------
// Per-bucket mean of fake (rows [0,nHalf)) and real (rows [nHalf,2*nHalf))
// halves of the feature matrix -> out (2*B, F)
// ---------------------------------------------------------------------------
__global__ __launch_bounds__(256)
void seg_avg_kernel(const float* __restrict__ feats, const int* __restrict__ counts,
                    const int* __restrict__ starts, float* __restrict__ out,
                    int nHalf, int F, int B)
{
    int b = blockIdx.x;
    int f = blockIdx.y * 256 + threadIdx.x;
    int cnt = counts[b];
    int st  = starts[b];
    float sf = 0.0f, sr = 0.0f;
    for (int i = 0; i < cnt; ++i) {
        long long r = st + i;
        sf += feats[r * (long long)F + f];
        sr += feats[(r + nHalf) * (long long)F + f];
    }
    float inv = (cnt > 0) ? (1.0f / (float)cnt) : 0.0f;
    out[(long long)b * F + f]       = sf * inv;
    out[(long long)(B + b) * F + f] = sr * inv;
}

// ---------------------------------------------------------------------------
// Launch
// ---------------------------------------------------------------------------
extern "C" void kernel_launch(void* const* d_in, const int* in_sizes, int n_in,
                              void* d_out, int out_size, void* d_ws, size_t ws_size,
                              hipStream_t stream)
{
    const float* x  = (const float*)d_in[0];
    const float* W1 = (const float*)d_in[1];
    const float* b1 = (const float*)d_in[2];
    const float* W2 = (const float*)d_in[3];
    const float* b2 = (const float*)d_in[4];
    const float* W3 = (const float*)d_in[5];
    const float* b3 = (const float*)d_in[6];
    const float* W4 = (const float*)d_in[7];
    const float* b4 = (const float*)d_in[8];
    const int*  seg = (const int*)d_in[9];

    const int Nimg = in_sizes[0] / (3 * 64 * 64);   // 4096
    const int nObj = in_sizes[9];                   // 2048
    const int F    = 512 * 4 * 4;                   // 8192
    const int B    = out_size / (2 * F);            // 128

    // K sizes (Cin*25) and padded variants
    const int K1 = 3 * 25,   K1p = (K1 + 31) & ~31;   // 75 -> 96
    const int K2 = 64 * 25,  K2p = K2;                // 1600
    const int K3 = 128 * 25, K3p = K3;                // 3200
    const int K4 = 256 * 25, K4p = K4;                // 6400

    // workspace layout:
    //   region A: y1(bf16) then y3(bf16)     region B: y2(bf16) then y4(f32)
    //   region W: padded bf16 weights        region C: counts/starts
    char* ws = (char*)d_ws;
    size_t szA = (size_t)Nimg * 64 * 32 * 32 * sizeof(unsigned short); // y1
    size_t szB = (size_t)Nimg * 128 * 16 * 16 * sizeof(unsigned short);// y2 (>= y4 bytes)
    unsigned short* y1 = (unsigned short*)(ws);
    unsigned short* y2 = (unsigned short*)(ws + szA);
    unsigned short* y3 = (unsigned short*)(ws);        // reuses y1 region
    float*          y4 = (float*)(ws + szA);           // reuses y2 region

    unsigned short* w1b = (unsigned short*)(ws + szA + szB);
    unsigned short* w2b = w1b + (size_t)64  * K1p;
    unsigned short* w3b = w2b + (size_t)128 * K2p;
    unsigned short* w4b = w3b + (size_t)256 * K3p;
    int* counts = (int*)(w4b + (size_t)512 * K4p);
    int* starts = counts + B;

    // ---- stage 0: weights -> padded bf16 (L2-resident, ~8.6MB) ----
    {
        int t1 = 64 * K1p, t2 = 128 * K2p, t3 = 256 * K3p, t4 = 512 * K4p;
        convert_weights_kernel<<<(t1 + 255) / 256, 256, 0, stream>>>(W1, w1b, K1, K1p, t1);
        convert_weights_kernel<<<(t2 + 255) / 256, 256, 0, stream>>>(W2, w2b, K2, K2p, t2);
        convert_weights_kernel<<<(t3 + 255) / 256, 256, 0, stream>>>(W3, w3b, K3, K3p, t3);
        convert_weights_kernel<<<(t4 + 255) / 256, 256, 0, stream>>>(W4, w4b, K4, K4p, t4);
    }

    const int M1 = Nimg * 32 * 32;
    const int M2 = Nimg * 16 * 16;
    const int M3 = Nimg * 8 * 8;
    const int M4 = Nimg * 4 * 4;

    // grid.y = Cout / (64*NSUB): full-Cout blocks -> activations read ~once
    conv5x5s2_lrelu_wmma<float, unsigned short, 1, 96>
        <<<dim3(M1 / 64, 1), 128, 0, stream>>>(x, w1b, b1, y1,
                                               3, 64, 64, 64, 32, 32, 32 * 32, K1, K1p);
    conv5x5s2_lrelu_wmma<unsigned short, unsigned short, 2, 1600>
        <<<dim3(M2 / 64, 1), 128, 0, stream>>>(y1, w2b, b2, y2,
                                               64, 32, 32, 128, 16, 16, 16 * 16, K2, K2p);
    conv5x5s2_lrelu_wmma<unsigned short, unsigned short, 4, 3200>
        <<<dim3(M3 / 64, 1), 128, 0, stream>>>(y2, w3b, b3, y3,
                                               128, 16, 16, 256, 8, 8, 8 * 8, K3, K3p);
    conv5x5s2_lrelu_wmma<unsigned short, float, 4, 6400>
        <<<dim3(M4 / 64, 2), 128, 0, stream>>>(y3, w4b, b4, y4,
                                               256, 8, 8, 512, 4, 4, 4 * 4, K4, K4p);

    seg_count_kernel<<<1, 128, 0, stream>>>(seg, nObj, B, counts, starts);
    seg_avg_kernel<<<dim3(B, F / 256), 256, 0, stream>>>(y4, counts, starts,
                                                         (float*)d_out, nObj, F, B);
}
